// Attention_69020124447023
// MI455X (gfx1250) — matrix operand order repaired and checked by reading, not compiled, not run on hardware
//
#include <hip/hip_runtime.h>

typedef __attribute__((ext_vector_type(16))) __bf16 v16bf;
typedef __attribute__((ext_vector_type(8)))  float  v8f;
typedef unsigned short u16t;

union BF16x16 {
  u16t  h[16];
  uint4 q[2];
  v16bf v;
};

__device__ __forceinline__ u16t f2bf(float f) {
  unsigned u = __float_as_uint(f);
  u += 0x7FFFu + ((u >> 16) & 1u);   // round-to-nearest-even
  return (u16t)(u >> 16);
}

__device__ __forceinline__ v8f v8zero() {
  v8f z = {0.f, 0.f, 0.f, 0.f, 0.f, 0.f, 0.f, 0.f};
  return z;
}

__device__ __forceinline__ v8f wmma_bf16(v16bf a, v16bf b, v8f c) {
  // D = A(16x32 bf16) * B(32x16 bf16) + C(16x16 f32)
  return __builtin_amdgcn_wmma_f32_16x16x32_bf16(false, a, false, b, (short)0, c,
                                                 false, false);
}

// A fragment 16x32 (MxK), row-major source with leading dim `ld` (elements).
// ISA layout: lane<16 holds row=lane, K = {h*8..h*8+7, 16+h*8..16+h*8+7}, h=lane>>4.
__device__ __forceinline__ v16bf load_a(const u16t* base, int rowBase, int ld,
                                        int k0, int lane) {
  const int nl = lane & 15, hf = lane >> 4;
  const u16t* p = base + (size_t)(rowBase + nl) * ld + k0 + hf * 8;
  BF16x16 f;
  f.q[0] = *(const uint4*)(p);
  f.q[1] = *(const uint4*)(p + 16);
  return f.v;
}

// B fragment 32x16 (KxN) from N-major storage (row = output column, K contiguous).
// ISA layout: lane n = column n&15; K = {h*8..h*8+7, 16+h*8..16+h*8+7}, h=lane>>4.
__device__ __forceinline__ v16bf load_b(const u16t* base, int colBase, int ld,
                                        int k0, int lane) {
  const int nl = lane & 15, hf = lane >> 4;
  const u16t* p = base + (size_t)(colBase + nl) * ld + k0 + hf * 8;
  BF16x16 f;
  f.q[0] = *(const uint4*)(p);
  f.q[1] = *(const uint4*)(p + 16);
  return f.v;
}

// ---------------------------------------------------------------- converts
__global__ void cvt_bf16(const float* __restrict__ s, u16t* __restrict__ d, int n) {
  int i = blockIdx.x * blockDim.x + threadIdx.x;
  const int stride = gridDim.x * blockDim.x;
  for (; i < n; i += stride) d[i] = f2bf(s[i]);
}

// src [R,C] row-major f32 -> dst [C,R] row-major bf16 (N-major weights for B frags)
__global__ void cvtT_bf16(const float* __restrict__ s, u16t* __restrict__ d,
                          int R, int C) {
  int i = blockIdx.x * blockDim.x + threadIdx.x;
  const int stride = gridDim.x * blockDim.x;
  const int n = R * C;
  for (; i < n; i += stride) {
    int r = i / C, c = i - r * C;
    d[(size_t)c * R + r] = f2bf(s[i]);
  }
}

// ---------------------------------------------------------------- QKV GEMM
// C[4096,3072] = X[4096,1024] * W (W stored [3072,1024] = [N,K]) + bias
// 64x64 tile per wave; epilogue scatters to Q/K [B,H,S,64] bf16 and V^T [B,H,64,S].
__global__ void __launch_bounds__(256) qkv_gemm_kernel(
    const u16t* __restrict__ X, const u16t* __restrict__ W,
    const float* __restrict__ bias, u16t* __restrict__ Qb,
    u16t* __restrict__ Kb, u16t* __restrict__ VTb) {
  const int lane = threadIdx.x & 31, wave = threadIdx.x >> 5;
  const int tile = blockIdx.x * 8 + wave;          // 3072 wave-tiles (64 x 48)
  const int tM = tile / 48, tN = tile - tM * 48;
  const int mBase = tM * 64, nBase = tN * 64;

  v8f acc[4][4];
#pragma unroll
  for (int i = 0; i < 4; ++i)
#pragma unroll
    for (int j = 0; j < 4; ++j) acc[i][j] = v8zero();

  for (int k0 = 0; k0 < 1024; k0 += 32) {
    v16bf a[4];
#pragma unroll
    for (int i = 0; i < 4; ++i) a[i] = load_a(X, mBase + i * 16, 1024, k0, lane);
#pragma unroll
    for (int j = 0; j < 4; ++j) {
      v16bf bf = load_b(W, nBase + j * 16, 1024, k0, lane);
#pragma unroll
      for (int i = 0; i < 4; ++i) acc[i][j] = wmma_bf16(a[i], bf, acc[i][j]);
    }
  }

  const int nl = lane & 15, hf = lane >> 4;
#pragma unroll
  for (int j = 0; j < 4; ++j) {
    const int col = nBase + j * 16 + nl;
    const float bv = bias[col];
    const int which = col >> 10;           // 0=Q 1=K 2=V
    const int within = col & 1023;
    const int h = within >> 6, d = within & 63;
#pragma unroll
    for (int i = 0; i < 4; ++i) {
#pragma unroll
      for (int r = 0; r < 8; ++r) {
        const int m = mBase + i * 16 + hf * 8 + r;   // global row = b*S+s
        const int bIdx = m >> 11, s = m & 2047;
        const int bh = bIdx * 16 + h;
        const u16t v = f2bf(acc[i][j][r] + bv);
        if (which == 0)      Qb[((size_t)bh * 2048 + s) * 64 + d] = v;
        else if (which == 1) Kb[((size_t)bh * 2048 + s) * 64 + d] = v;
        else                 VTb[((size_t)bh * 64 + d) * 2048 + s] = v;
      }
    }
  }
}

// ---------------------------------------------------------------- attention
// One wave per (b,h, 32-row q tile): 2 q-row blocks share K/V fragments so each
// 32-key chunk issues 16 WMMAs against 8KB of K/V loads (32 FLOP/B from L2).
__global__ void __launch_bounds__(128) attn_kernel(
    const u16t* __restrict__ Qb, const u16t* __restrict__ Kb,
    const u16t* __restrict__ VTb, u16t* __restrict__ CTX) {
  __shared__ u16t ldsP[4][32 * 32];   // per-wave P(32x32) scratch, C-layout -> A-layout
  const int lane = threadIdx.x & 31, wave = threadIdx.x >> 5;
  const int wg = blockIdx.x * 4 + wave;            // 2048 waves total
  const int qt = wg & 63, bh = wg >> 6;            // bh = b*16+h
  const int b = bh >> 4, h = bh & 15;
  const int qBase = qt * 32;
  const u16t* qp = Qb + (size_t)bh * 2048 * 64;
  const u16t* kp = Kb + (size_t)bh * 2048 * 64;
  const u16t* vp = VTb + (size_t)bh * 64 * 2048;
  const int nl = lane & 15, hf = lane >> 4;

  v16bf qa[2][2];
#pragma unroll
  for (int ib = 0; ib < 2; ++ib) {
    qa[ib][0] = load_a(qp, qBase + ib * 16, 64, 0, lane);    // d = 0..31
    qa[ib][1] = load_a(qp, qBase + ib * 16, 64, 32, lane);   // d = 32..63
  }

  float mrun[2][8], lrun[2][8];
#pragma unroll
  for (int ib = 0; ib < 2; ++ib)
#pragma unroll
    for (int r = 0; r < 8; ++r) { mrun[ib][r] = -1e30f; lrun[ib][r] = 0.f; }
  v8f oacc[2][4];
#pragma unroll
  for (int ib = 0; ib < 2; ++ib)
#pragma unroll
    for (int j = 0; j < 4; ++j) oacc[ib][j] = v8zero();

  u16t* lp = &ldsP[wave][0];
  const int kEnd = qBase + 32;                      // causal: keys < kEnd matter
  for (int kb = 0; kb < kEnd; kb += 32) {
    // K fragments for this key chunk (shared by both q row-blocks)
    v16bf kf[2][2];
#pragma unroll
    for (int nb = 0; nb < 2; ++nb) {
      kf[nb][0] = load_b(kp, kb + nb * 16, 64, 0, lane);
      kf[nb][1] = load_b(kp, kb + nb * 16, 64, 32, lane);
    }
    // scores: 4 C tiles (2 q row-blocks x 2 key blocks), 8 WMMAs
    v8f sc[2][2];
#pragma unroll
    for (int ib = 0; ib < 2; ++ib)
#pragma unroll
      for (int nb = 0; nb < 2; ++nb) {
        v8f s = v8zero();
        s = wmma_bf16(qa[ib][0], kf[nb][0], s);
        s = wmma_bf16(qa[ib][1], kf[nb][1], s);
        sc[ib][nb] = s;
      }

    // online softmax per q row-block
#pragma unroll
    for (int ib = 0; ib < 2; ++ib) {
      float p0[8], p1[8], rm[8];
#pragma unroll
      for (int r = 0; r < 8; ++r) {
        const int q = qBase + ib * 16 + hf * 8 + r;
        float v0 = sc[ib][0][r] * 0.125f;              // 1/sqrt(64)
        if (kb + nl > q) v0 -= 10000.f;                // reference's additive mask
        float v1 = sc[ib][1][r] * 0.125f;
        if (kb + 16 + nl > q) v1 -= 10000.f;
        p0[r] = v0; p1[r] = v1;
        rm[r] = fmaxf(v0, v1);
      }
#pragma unroll
      for (int r = 0; r < 8; ++r) {
#pragma unroll
        for (int msk = 1; msk < 16; msk <<= 1)
          rm[r] = fmaxf(rm[r], __shfl_xor(rm[r], msk, 32));
      }
      float rs[8];
#pragma unroll
      for (int r = 0; r < 8; ++r) {
        const float mn = fmaxf(mrun[ib][r], rm[r]);
        const float alpha = __expf(mrun[ib][r] - mn);
        const float e0 = __expf(p0[r] - mn);
        const float e1 = __expf(p1[r] - mn);
        mrun[ib][r] = mn;
        p0[r] = e0; p1[r] = e1;
        rs[r] = e0 + e1;
        lrun[ib][r] = lrun[ib][r] * alpha;
#pragma unroll
        for (int j = 0; j < 4; ++j) oacc[ib][j][r] = oacc[ib][j][r] * alpha;
      }
#pragma unroll
      for (int r = 0; r < 8; ++r) {
#pragma unroll
        for (int msk = 1; msk < 16; msk <<= 1)
          rs[r] += __shfl_xor(rs[r], msk, 32);
        lrun[ib][r] += rs[r];
      }
      // spill P row-block (16x32) to LDS rows ib*16..
#pragma unroll
      for (int r = 0; r < 8; ++r) {
        const int row = ib * 16 + hf * 8 + r;
        lp[row * 32 + nl]      = f2bf(p0[r]);
        lp[row * 32 + 16 + nl] = f2bf(p1[r]);
      }
    }
    asm volatile("s_wait_dscnt 0" ::: "memory");   // intra-wave cross-lane LDS fence

    // P(32x32) * V(32x64): V fragments shared by both q row-blocks, 8 WMMAs
    v16bf pa[2];
    pa[0] = load_a(lp, 0, 32, 0, lane);
    pa[1] = load_a(lp, 16, 32, 0, lane);
#pragma unroll
    for (int j = 0; j < 4; ++j) {
      v16bf vf = load_b(vp, j * 16, 2048, kb, lane);
      oacc[0][j] = wmma_bf16(pa[0], vf, oacc[0][j]);
      oacc[1][j] = wmma_bf16(pa[1], vf, oacc[1][j]);
    }
  }

  // normalize and write ctx (merged heads, bf16) -> [B,S,D]
#pragma unroll
  for (int ib = 0; ib < 2; ++ib)
#pragma unroll
    for (int j = 0; j < 4; ++j) {
#pragma unroll
      for (int r = 0; r < 8; ++r) {
        const int q = qBase + ib * 16 + hf * 8 + r;
        const float o = oacc[ib][j][r] / lrun[ib][r];
        CTX[((size_t)(b * 2048 + q)) * 1024 + h * 64 + j * 16 + nl] = f2bf(o);
      }
    }
}

// ---------------------------------------------------------------- out proj
// out[4096,1024] = CTX[4096,1024] * W (stored [1024,1024] = [N,K]) + bias, fp32 out
__global__ void __launch_bounds__(256) proj_gemm_kernel(
    const u16t* __restrict__ CTX, const u16t* __restrict__ W,
    const float* __restrict__ bias, float* __restrict__ out) {
  const int lane = threadIdx.x & 31, wave = threadIdx.x >> 5;
  const int tile = blockIdx.x * 8 + wave;          // 1024 wave-tiles (64 x 16)
  const int tM = tile >> 4, tN = tile & 15;
  const int mBase = tM * 64, nBase = tN * 64;

  v8f acc[4][4];
#pragma unroll
  for (int i = 0; i < 4; ++i)
#pragma unroll
    for (int j = 0; j < 4; ++j) acc[i][j] = v8zero();

  for (int k0 = 0; k0 < 1024; k0 += 32) {
    v16bf a[4];
#pragma unroll
    for (int i = 0; i < 4; ++i) a[i] = load_a(CTX, mBase + i * 16, 1024, k0, lane);
#pragma unroll
    for (int j = 0; j < 4; ++j) {
      v16bf bf = load_b(W, nBase + j * 16, 1024, k0, lane);
#pragma unroll
      for (int i = 0; i < 4; ++i) acc[i][j] = wmma_bf16(a[i], bf, acc[i][j]);
    }
  }

  const int nl = lane & 15, hf = lane >> 4;
#pragma unroll
  for (int j = 0; j < 4; ++j) {
    const int col = nBase + j * 16 + nl;
    const float bv = bias[col];
#pragma unroll
    for (int i = 0; i < 4; ++i) {
#pragma unroll
      for (int r = 0; r < 8; ++r) {
        const int m = mBase + i * 16 + hf * 8 + r;
        out[(size_t)m * 1024 + col] = acc[i][j][r] + bv;
      }
    }
  }
}

// ---------------------------------------------------------------- launch
extern "C" void kernel_launch(void* const* d_in, const int* in_sizes, int n_in,
                              void* d_out, int out_size, void* d_ws, size_t ws_size,
                              hipStream_t stream) {
  (void)in_sizes; (void)n_in; (void)out_size; (void)ws_size;
  const float* hs  = (const float*)d_in[0];   // [2,2048,1024]
  const float* caw = (const float*)d_in[1];   // [1024,3072]
  const float* cab = (const float*)d_in[2];   // [3072]
  const float* cpw = (const float*)d_in[3];   // [1024,1024]
  const float* cpb = (const float*)d_in[4];   // [1024]
  // d_in[5] causal mask unused: causality computed analytically.

  char* ws = (char*)d_ws;
  u16t* X      = (u16t*)(ws);                        // 8 MB  [4096,1024] bf16
  u16t* WqkvT  = (u16t*)(ws + ((size_t)8  << 20));   // 6 MB  [3072,1024] bf16
  u16t* WprojT = (u16t*)(ws + ((size_t)14 << 20));   // 2 MB  [1024,1024] bf16
  u16t* Qb     = (u16t*)(ws + ((size_t)16 << 20));   // 8 MB  [32,2048,64] bf16
  u16t* Kb     = (u16t*)(ws + ((size_t)24 << 20));   // 8 MB  [32,2048,64] bf16
  u16t* VTb    = (u16t*)(ws + ((size_t)32 << 20));   // 8 MB  [32,64,2048] bf16
  u16t* CTXb   = (u16t*)(ws + ((size_t)40 << 20));   // 8 MB  [4096,1024] bf16

  cvt_bf16 <<<2048, 256, 0, stream>>>(hs, X, 4096 * 1024);
  cvtT_bf16<<<2048, 256, 0, stream>>>(caw, WqkvT, 1024, 3072);
  cvtT_bf16<<<1024, 256, 0, stream>>>(cpw, WprojT, 1024, 1024);
  qkv_gemm_kernel<<<384, 256, 0, stream>>>(X, WqkvT, cab, Qb, Kb, VTb);
  attn_kernel    <<<512, 128, 0, stream>>>(Qb, Kb, VTb, CTXb);
  proj_gemm_kernel<<<128, 256, 0, stream>>>(CTXb, WprojT, cpb, (float*)d_out);
}
